// Decoder_ODE_RNN_cluster_14319420965185
// MI455X (gfx1250) — compile-verified, matrix operand order checked by reading, not run on hardware
//
#include <hip/hip_runtime.h>
#include <hip/hip_bf16.h>
#include <math.h>

// ---------------- types ----------------
typedef __bf16 bf16_t;
typedef __attribute__((ext_vector_type(16))) __bf16 v16bf;
typedef __attribute__((ext_vector_type(8)))  float  v8f;
typedef __attribute__((ext_vector_type(8)))  unsigned short u16x8; // 16-byte chunk

// ---------------- problem dims ----------------
#define LATENT  1024
#define INPUT   512
#define CLUSTER 256
#define NUNITS  100
#define NUP     128       // n_units padded to tile multiple
#define NTRAJ   512
#define NTP     64
#define NSTEPS  (NTP - 1) // 63
#define DCAT    (LATENT + INPUT) // 1536

// epilogue modes
#define EP_SIGMOID   0
#define EP_GRU       1
#define EP_TANH_BF16 2
#define EP_EULER     3
#define EP_PLAIN     4

// ---- CDNA5 async copy: memory -> LDS, tracked by ASYNCcnt ----
#define ASYNC_CP_B128(ldsaddr, gptr)                                        \
    asm volatile("global_load_async_to_lds_b128 %0, %1, off"                \
                 :: "v"(ldsaddr),                                           \
                    "v"((unsigned long long)(size_t)(gptr))                 \
                 : "memory")
#define WAIT_ASYNC0() asm volatile("s_wait_asynccnt 0x0" ::: "memory")

__device__ __forceinline__ unsigned lds_u32(const void* p)
{
    // flat->LDS mapping truncates to addr[31:0] (ISA 10.2): low 32 bits of
    // the generic pointer into __shared__ are the LDS byte address.
    return (unsigned)(size_t)p;
}

// =====================================================================
// Generic bf16 WMMA GEMM:  C[M,N] = A[M,K] @ B^T[N,K]  (B stored [N][K])
// Block tile 128(M) x 128(N), K-step 32, 256 threads = 8 waves (4x2).
// Each wave computes a 32x64 patch = 2x4 v_wmma_f32_16x16x32_bf16 tiles.
// Double-buffered LDS (2 x 16KB) filled by GLOBAL_LOAD_ASYNC_TO_LDS_B128
// (no VGPR staging), synchronized via ASYNCcnt; k+2 tile prefetched to L2.
// =====================================================================
__global__ __launch_bounds__(256)
void gemm_bf16_wmma(const bf16_t* __restrict__ A, int lda,
                    const bf16_t* __restrict__ B, int ldb,
                    const float*  __restrict__ bias, int nbias,
                    int K,
                    float*  __restrict__ outF,  int ldoF,
                    bf16_t* __restrict__ outB1, int ldoB1,
                    bf16_t* __restrict__ outB2, int ldoB2,
                    const float* __restrict__ U,     // gate u [512,1024]
                    const float* __restrict__ Yold,  // state  [512,1024]
                    float scale, int mode)
{
    __shared__ __align__(16) bf16_t la[2][128 * 32];
    __shared__ __align__(16) bf16_t lb[2][128 * 32];

    const int tid  = threadIdx.x;
    const int lane = tid & 31;
    const int wave = tid >> 5;
    const int wm   = wave & 3;   // M-wave index 0..3 -> rows wm*32
    const int wn   = wave >> 2;  // N-wave index 0..1 -> cols wn*64
    const int rowBase = blockIdx.y * 128;
    const int colBase = blockIdx.x * 128;

    const int r = lane & 15;     // row/col within 16
    const int h = lane >> 4;     // half select (K grouping)

    // per-thread staging chunk coordinates (512 chunks of 16B per tile)
    const int c0row = tid >> 2,           c0col = (tid & 3) * 8;
    const int c1row = (tid + 256) >> 2,   c1col = ((tid + 256) & 3) * 8;

    // LDS byte addresses of this thread's four 16B destinations, per buffer
    unsigned dA0[2], dA1[2], dB0[2], dB1[2];
    #pragma unroll
    for (int b = 0; b < 2; ++b) {
        dA0[b] = lds_u32(la[b] + c0row * 32 + c0col);
        dA1[b] = lds_u32(la[b] + c1row * 32 + c1col);
        dB0[b] = lds_u32(lb[b] + c0row * 32 + c0col);
        dB1[b] = lds_u32(lb[b] + c1row * 32 + c1col);
    }
    // global base addresses of this thread's chunks (advance by k)
    const bf16_t* gA0 = A + (size_t)(rowBase + c0row) * lda + c0col;
    const bf16_t* gA1 = A + (size_t)(rowBase + c1row) * lda + c1col;
    const bf16_t* gB0 = B + (size_t)(colBase + c0row) * ldb + c0col;
    const bf16_t* gB1 = B + (size_t)(colBase + c1row) * ldb + c1col;

    const int nk = K >> 5;
    v8f acc[2][4] = {};

    // ---- prologue: async-copy K-tile 0 into buffer 0 ----
    ASYNC_CP_B128(dA0[0], gA0);
    ASYNC_CP_B128(dA1[0], gA1);
    ASYNC_CP_B128(dB0[0], gB0);
    ASYNC_CP_B128(dB1[0], gB1);
    WAIT_ASYNC0();
    __syncthreads();

    for (int kt = 0; kt < nk; ++kt) {
        const int cur = kt & 1;

        // ---- async-copy K-tile kt+1 into the back buffer ----
        if (kt + 1 < nk) {
            const int k1 = (kt + 1) << 5;
            ASYNC_CP_B128(dA0[cur ^ 1], gA0 + k1);
            ASYNC_CP_B128(dA1[cur ^ 1], gA1 + k1);
            ASYNC_CP_B128(dB0[cur ^ 1], gB0 + k1);
            ASYNC_CP_B128(dB1[cur ^ 1], gB1 + k1);
        }
        // ---- prefetch K-tile kt+2 into L2 (global_prefetch_b8) ----
        if (kt + 2 < nk) {
            const int k2 = (kt + 2) << 5;
            __builtin_prefetch(gA0 + k2, 0, 1);
            __builtin_prefetch(gB0 + k2, 0, 1);
        }

        // ---- fragments from LDS per documented 16-bit A layout ----
        v16bf afrag[2], bfrag[4];
        #pragma unroll
        for (int i = 0; i < 2; ++i) {
            const bf16_t* p = la[cur] + (wm * 32 + i * 16 + r) * 32 + h * 8;
            union { v16bf v; u16x8 q[2]; } t;
            t.q[0] = *(const u16x8*)(p);
            t.q[1] = *(const u16x8*)(p + 16);
            afrag[i] = t.v;
        }
        #pragma unroll
        for (int j = 0; j < 4; ++j) {
            const bf16_t* p = lb[cur] + (wn * 64 + j * 16 + r) * 32 + h * 8;
            union { v16bf v; u16x8 q[2]; } t;
            t.q[0] = *(const u16x8*)(p);
            t.q[1] = *(const u16x8*)(p + 16);
            bfrag[j] = t.v;
        }

        #pragma unroll
        for (int i = 0; i < 2; ++i)
            #pragma unroll
            for (int j = 0; j < 4; ++j)
                acc[i][j] = __builtin_amdgcn_wmma_f32_16x16x32_bf16(
                    false, afrag[i], false, bfrag[j],
                    (short)0, acc[i][j], false, false);

        // ---- own async copies done, then block-wide flip ----
        WAIT_ASYNC0();
        __syncthreads();
    }

    // ---- fused epilogue ----
    #pragma unroll
    for (int i = 0; i < 2; ++i) {
        #pragma unroll
        for (int j = 0; j < 4; ++j) {
            int col = colBase + wn * 64 + j * 16 + (lane & 15);
            float bv = (col < nbias) ? bias[col] : 0.0f;
            #pragma unroll
            for (int v = 0; v < 8; ++v) {
                int row = rowBase + wm * 32 + i * 16 + (lane >> 4) * 8 + v;
                float a = acc[i][j][v] + bv;
                switch (mode) {
                case EP_SIGMOID: {
                    outF[(size_t)row * ldoF + col] = 1.0f / (1.0f + __expf(-a));
                } break;
                case EP_PLAIN: {
                    outF[(size_t)row * ldoF + col] = a;
                } break;
                case EP_TANH_BF16: {
                    outB1[(size_t)row * ldoB1 + col] = (bf16_t)tanhf(a);
                } break;
                case EP_GRU: {
                    float u  = U[(size_t)row * LATENT + col];
                    float yo = Yold[(size_t)row * LATENT + col];
                    float yn = (1.0f - u) * a + u * yo;
                    outF[(size_t)row * ldoF + col]   = yn;         // Y fp32
                    outB1[(size_t)row * ldoB1 + col] = (bf16_t)yn; // Ybf for ODE
                } break;
                case EP_EULER: {
                    float yo = Yold[(size_t)row * LATENT + col];
                    float yn = yo + scale * a;
                    outF[(size_t)row * ldoF + col]   = yn;         // Y fp32
                    outB1[(size_t)row * ldoB1 + col] = (bf16_t)yn; // Ybf
                    if (outB2)
                        outB2[(size_t)row * ldoB2 + col] = (bf16_t)yn; // concat buf
                } break;
                }
            }
        }
    }
}

// =====================================================================
// Support kernels
// =====================================================================

// transpose + cvt: in [K][N] fp32 -> out [Nout][Kout] bf16, zero-padded
__global__ void tconv_kernel(const float* __restrict__ in, bf16_t* __restrict__ out,
                             int K, int N, int Kout, int Nout)
{
    size_t idx = (size_t)blockIdx.x * blockDim.x + threadIdx.x;
    if (idx >= (size_t)Nout * Kout) return;
    int n = (int)(idx / Kout);
    int k = (int)(idx % Kout);
    float v = (n < N && k < K) ? in[(size_t)k * N + n] : 0.0f;
    out[idx] = (bf16_t)v;
}

// zero Y (fp32) and the Y-part of the concat bf16 buffer
__global__ void init_state_kernel(float* __restrict__ Y, bf16_t* __restrict__ Abf)
{
    int idx = blockIdx.x * blockDim.x + threadIdx.x; // 512*1024
    int row = idx >> 10, col = idx & 1023;
    Y[idx] = 0.0f;
    Abf[(size_t)row * DCAT + col] = (bf16_t)0.0f;
}

// Abf[:, 1024:1536] = bf16(data[:, t, :])
__global__ void conv_x_kernel(const float* __restrict__ data,
                              bf16_t* __restrict__ Abf, int t)
{
    int idx = blockIdx.x * blockDim.x + threadIdx.x; // 512*512
    int row = idx >> 9, col = idx & 511;
    float v = data[((size_t)row * NTP + t) * INPUT + col];
    Abf[(size_t)row * DCAT + LATENT + col] = (bf16_t)v;
}

// Abf[:, 0:1024] = bf16(Y * r)
__global__ void conv_yr_kernel(const float* __restrict__ Y,
                               const float* __restrict__ R,
                               bf16_t* __restrict__ Abf)
{
    int idx = blockIdx.x * blockDim.x + threadIdx.x; // 512*1024
    int row = idx >> 10, col = idx & 1023;
    Abf[(size_t)row * DCAT + col] = (bf16_t)(Y[idx] * R[idx]);
}

// row-wise softmax over 256 columns
__global__ void softmax_kernel(const float* __restrict__ logits,
                               float* __restrict__ out)
{
    __shared__ float red[256];
    int row = blockIdx.x;
    int c   = threadIdx.x;
    float v = logits[(size_t)row * CLUSTER + c];
    red[c] = v; __syncthreads();
    for (int s = 128; s > 0; s >>= 1) {
        if (c < s) red[c] = fmaxf(red[c], red[c + s]);
        __syncthreads();
    }
    float m = red[0]; __syncthreads();
    float e = __expf(v - m);
    red[c] = e; __syncthreads();
    for (int s = 128; s > 0; s >>= 1) {
        if (c < s) red[c] += red[c + s];
        __syncthreads();
    }
    out[(size_t)row * CLUSTER + c] = e / red[0];
}

__global__ void copy_f32_kernel(const float* __restrict__ src,
                                float* __restrict__ dst)
{
    int idx = blockIdx.x * blockDim.x + threadIdx.x; // 512*1024
    dst[idx] = src[idx];
}

// =====================================================================
// Host orchestration
// =====================================================================
extern "C" void kernel_launch(void* const* d_in, const int* in_sizes, int n_in,
                              void* d_out, int out_size, void* d_ws, size_t ws_size,
                              hipStream_t stream)
{
    (void)in_sizes; (void)n_in; (void)out_size; (void)ws_size;
    const float* data = (const float*)d_in[0];
    // d_in[1] = time_steps (arange => dt = 1.0, sub = 0.5)
    const float* W_u = (const float*)d_in[2];  const float* b_u = (const float*)d_in[3];
    const float* W_r = (const float*)d_in[4];  const float* b_r = (const float*)d_in[5];
    const float* W_n = (const float*)d_in[6];  const float* b_n = (const float*)d_in[7];
    const float* W1  = (const float*)d_in[8];  const float* b1  = (const float*)d_in[9];
    const float* W2  = (const float*)d_in[10]; const float* b2  = (const float*)d_in[11];
    const float* We  = (const float*)d_in[12]; const float* be  = (const float*)d_in[13];
    float* out = (float*)d_out;

    // ---- workspace carve-up (bytes, 256-aligned) ----
    char* p = (char*)d_ws;
    auto carve = [&](size_t bytes) -> char* {
        char* q = p;
        p += (bytes + 255) & ~(size_t)255;
        return q;
    };
    bf16_t* WuT = (bf16_t*)carve((size_t)LATENT * DCAT * 2);   // [1024][1536]
    bf16_t* WrT = (bf16_t*)carve((size_t)LATENT * DCAT * 2);
    bf16_t* WnT = (bf16_t*)carve((size_t)LATENT * DCAT * 2);
    bf16_t* W1T = (bf16_t*)carve((size_t)NUP * LATENT * 2);    // [128][1024] (pad)
    bf16_t* W2T = (bf16_t*)carve((size_t)LATENT * NUP * 2);    // [1024][128] (pad)
    bf16_t* WeT = (bf16_t*)carve((size_t)CLUSTER * LATENT * 2);// [256][1024]
    bf16_t* Abf = (bf16_t*)carve((size_t)NTRAJ * DCAT * 2);    // [512][1536] concat
    bf16_t* Ybf = (bf16_t*)carve((size_t)NTRAJ * LATENT * 2);  // [512][1024]
    bf16_t* H1  = (bf16_t*)carve((size_t)NTRAJ * NUP * 2);     // [512][128]
    float*  Y   = (float*) carve((size_t)NTRAJ * LATENT * 4);
    float*  Ubuf= (float*) carve((size_t)NTRAJ * LATENT * 4);
    float*  Rbuf= (float*) carve((size_t)NTRAJ * LATENT * 4);
    float*  Lg  = (float*) carve((size_t)NTRAJ * CLUSTER * 4);

    // ---- one-time weight convert + transpose (bf16, B stored [N][K]) ----
    tconv_kernel<<<(LATENT * DCAT + 255) / 256, 256, 0, stream>>>(W_u, WuT, DCAT, LATENT, DCAT, LATENT);
    tconv_kernel<<<(LATENT * DCAT + 255) / 256, 256, 0, stream>>>(W_r, WrT, DCAT, LATENT, DCAT, LATENT);
    tconv_kernel<<<(LATENT * DCAT + 255) / 256, 256, 0, stream>>>(W_n, WnT, DCAT, LATENT, DCAT, LATENT);
    tconv_kernel<<<(NUP * LATENT + 255) / 256, 256, 0, stream>>>(W1, W1T, LATENT, NUNITS, LATENT, NUP);
    tconv_kernel<<<(LATENT * NUP + 255) / 256, 256, 0, stream>>>(W2, W2T, NUNITS, LATENT, NUP, LATENT);
    tconv_kernel<<<(CLUSTER * LATENT + 255) / 256, 256, 0, stream>>>(We, WeT, LATENT, CLUSTER, LATENT, CLUSTER);
    init_state_kernel<<<(NTRAJ * LATENT) / 256, 256, 0, stream>>>(Y, Abf);

    auto gemm = [&](const bf16_t* A, int lda, const bf16_t* B, int ldb,
                    const float* bias, int nbias, int M, int N, int K,
                    float* oF, int ldoF, bf16_t* oB1, int ldoB1,
                    bf16_t* oB2, int ldoB2,
                    const float* U, const float* Yold, float scale, int mode) {
        dim3 grid(N / 128, M / 128);
        gemm_bf16_wmma<<<grid, 256, 0, stream>>>(A, lda, B, ldb, bias, nbias, K,
                                                 oF, ldoF, oB1, ldoB1, oB2, ldoB2,
                                                 U, Yold, scale, mode);
    };

    const size_t ys_elems = (size_t)NSTEPS * NTRAJ * CLUSTER;

    for (int t = 0; t < NSTEPS; ++t) {
        // x_t -> concat buffer
        conv_x_kernel<<<(NTRAJ * INPUT) / 256, 256, 0, stream>>>(data, Abf, t);

        // gates: u = sig([y|x] Wu + bu), r = sig([y|x] Wr + br)
        gemm(Abf, DCAT, WuT, DCAT, b_u, LATENT, NTRAJ, LATENT, DCAT,
             Ubuf, LATENT, nullptr, 0, nullptr, 0, nullptr, nullptr, 0.f, EP_SIGMOID);
        gemm(Abf, DCAT, WrT, DCAT, b_r, LATENT, NTRAJ, LATENT, DCAT,
             Rbuf, LATENT, nullptr, 0, nullptr, 0, nullptr, nullptr, 0.f, EP_SIGMOID);

        // n = [y*r|x] Wn + bn ; y = (1-u)n + u y  (writes Y fp32 + Ybf bf16)
        conv_yr_kernel<<<(NTRAJ * LATENT) / 256, 256, 0, stream>>>(Y, Rbuf, Abf);
        gemm(Abf, DCAT, WnT, DCAT, b_n, LATENT, NTRAJ, LATENT, DCAT,
             Y, LATENT, Ybf, LATENT, nullptr, 0, Ubuf, Y, 0.f, EP_GRU);

        // two Euler steps: y += 0.5 * (tanh(y W1 + b1) W2 + b2)
        for (int e = 0; e < 2; ++e) {
            gemm(Ybf, LATENT, W1T, LATENT, b1, NUNITS, NTRAJ, NUP, LATENT,
                 nullptr, 0, H1, NUP, nullptr, 0, nullptr, nullptr, 0.f, EP_TANH_BF16);
            gemm(H1, NUP, W2T, NUP, b2, LATENT, NTRAJ, LATENT, NUP,
                 Y, LATENT, Ybf, LATENT, Abf, DCAT, nullptr, Y, 0.5f, EP_EULER);
        }

        // emission: softmax(y We + be)
        gemm(Ybf, LATENT, WeT, LATENT, be, CLUSTER, NTRAJ, CLUSTER, LATENT,
             Lg, CLUSTER, nullptr, 0, nullptr, 0, nullptr, nullptr, 0.f, EP_PLAIN);
        softmax_kernel<<<NTRAJ, CLUSTER, 0, stream>>>(Lg, out + (size_t)t * NTRAJ * CLUSTER);

        // record latent state
        copy_f32_kernel<<<(NTRAJ * LATENT) / 256, 256, 0, stream>>>(
            Y, out + ys_elems + (size_t)t * NTRAJ * LATENT);
    }
}
